// CausalSelfAttention_6347961663839
// MI455X (gfx1250) — compile-verified, matrix operand order checked by reading, not run on hardware
//
#include <hip/hip_runtime.h>
#include <hip/hip_bf16.h>
#include <math.h>

// ---------------------------------------------------------------------------
// CausalSelfAttention for MI455X (gfx1250, wave32, WMMA + async-LDS copies).
//   B=4, S=2048, H=1024.  All matmuls via v_wmma_f32_16x16x32_bf16.
//   Pipeline: [1] QKV = x @ W (bf16 out, Q pre-scaled by 1/sqrt(H))
//             [2] S = Q @ K^T  (causal block skip, async tile copies)
//             [3] row softmax  (fp32 math, bf16 probs, zeros above diag)
//             [4] O = P @ V    (k clipped at diagonal, async P copies)
//   All GEMM K-loops are double-buffered: next tile is fetched (async-to-LDS
//   or via registers where conversion/transpose is needed) while WMMAs run on
//   the current buffer; one barrier per K-step.
//   Scores (33.5 MB bf16) live in workspace and fit MI455X's 192 MB L2.
// ---------------------------------------------------------------------------

typedef __attribute__((ext_vector_type(16))) __bf16 v16bf;
typedef __attribute__((ext_vector_type(8)))  float  v8f;

#define Bv   4
#define Sv   2048
#define Hv   1024
#define H3v  3072
#define MTOT (Bv * Sv)      // 8192
#define TILE 128
#define KSTEP 32
#define LDT  40             // LDS row stride in bf16 (80B: 16B-aligned, bank-spread)

union U4pack { unsigned long long u; __bf16 h[4]; };
union U8pack { uint4 q; __bf16 h[8]; };
union FragU  { v16bf v; uint4 q[2]; };

// --- CDNA5 async global->LDS copy (16B per lane, ASYNCcnt-tracked) ---------
// GVS addressing: mem = SGPR_base + VGPR_offset; LDS dest from VGPR (low 32
// bits of the generic pointer are the LDS byte address).
__device__ inline void async_copy_b128(void* lds_ptr, const void* sbase,
                                       unsigned vbyte_off) {
    unsigned lds_off = (unsigned)(unsigned long long)lds_ptr;
    asm volatile("global_load_async_to_lds_b128 %0, %1, %2"
                 :: "v"(lds_off), "v"(vbyte_off), "s"(sbase)
                 : "memory");
}
__device__ inline void wait_async0() {
    asm volatile("s_wait_asynccnt 0x0" ::: "memory");
}

// Load one 16x32 bf16 WMMA operand fragment from a row-major LDS tile.
// ISA layout (16-bit A 16x32): lanes 0-15 -> K{0..7,16..23}, lanes 16-31 -> K{8..15,24..31}.
__device__ inline v16bf frag_ld(const __bf16* base, int lane) {
    const __bf16* p = base + (size_t)(lane & 15) * LDT;
    int k0 = (lane & 16) ? 8 : 0;
    FragU u;
    u.q[0] = *(const uint4*)(p + k0);
    u.q[1] = *(const uint4*)(p + 16 + k0);
    return u.v;
}

__device__ inline v8f wmma_bf16(v16bf a, v16bf b, v8f c) {
    return __builtin_amdgcn_wmma_f32_16x16x32_bf16(
        /*neg_a=*/false, a, /*neg_b=*/false, b,
        /*c_mod=*/(short)0, c, /*reuse_a=*/false, /*reuse_b=*/false);
}

// Shared 128x128-tile compute step: 2x4 accumulator tiles per wave, 8 WMMAs.
__device__ inline void mma_step(const __bf16* Atile, const __bf16* Btile,
                                int wm, int wn, int lane, v8f (&acc)[2][4]) {
    v16bf af[2], bfv[4];
#pragma unroll
    for (int i = 0; i < 2; ++i) af[i]  = frag_ld(Atile + (wm + i * 16) * LDT, lane);
#pragma unroll
    for (int j = 0; j < 4; ++j) bfv[j] = frag_ld(Btile + (wn + j * 16) * LDT, lane);
#pragma unroll
    for (int i = 0; i < 2; ++i)
#pragma unroll
        for (int j = 0; j < 4; ++j) acc[i][j] = wmma_bf16(af[i], bfv[j], acc[i][j]);
}

// ---------------------------------------------------------------------------
// Kernel 1: QKV projection.  M[8192,3072] = X[8192,1024] @ W[1024,3072].
// fp32 -> bf16 conversion forces register staging; double-buffered LDS.
// ---------------------------------------------------------------------------
__global__ __launch_bounds__(256) void qkv_gemm_kernel(
    const float* __restrict__ X, const float* __restrict__ W,
    __bf16* __restrict__ M) {
    __shared__ __bf16 As[2][TILE * LDT];   // [row][k]
    __shared__ __bf16 Bts[2][TILE * LDT];  // [n][k]
    const int tid  = threadIdx.x;
    const int lane = tid & 31, wave = tid >> 5;
    const int m0 = blockIdx.x * TILE;
    const int n0 = blockIdx.y * TILE;
    const int wm = (wave & 3) * 32;
    const int wn = (wave >> 2) * 64;

    // Per-thread fixed tile coordinates (k0-independent).
    int arow[4], ac4[4], bkr[4], bc4[4];
#pragma unroll
    for (int u = 0; u < 4; ++u) {
        int idx = u * 256 + tid;
        arow[u] = idx >> 3;  ac4[u] = (idx & 7) * 4;     // A: 8 float4 / row
        bkr[u]  = idx >> 5;  bc4[u] = (idx & 31) * 4;    // B: 32 float4 / row
    }

    float4 aReg[4], bReg[4];
    auto load_regs = [&](int k0) {
#pragma unroll
        for (int u = 0; u < 4; ++u) {
            aReg[u] = *(const float4*)(X + (size_t)(m0 + arow[u]) * Hv + k0 + ac4[u]);
            bReg[u] = *(const float4*)(W + (size_t)(k0 + bkr[u]) * H3v + n0 + bc4[u]);
        }
    };
    auto store_tiles = [&](int buf) {
#pragma unroll
        for (int u = 0; u < 4; ++u) {
            U4pack p;
            p.h[0] = (__bf16)aReg[u].x; p.h[1] = (__bf16)aReg[u].y;
            p.h[2] = (__bf16)aReg[u].z; p.h[3] = (__bf16)aReg[u].w;
            *(unsigned long long*)&As[buf][arow[u] * LDT + ac4[u]] = p.u;
            Bts[buf][(bc4[u] + 0) * LDT + bkr[u]] = (__bf16)bReg[u].x;
            Bts[buf][(bc4[u] + 1) * LDT + bkr[u]] = (__bf16)bReg[u].y;
            Bts[buf][(bc4[u] + 2) * LDT + bkr[u]] = (__bf16)bReg[u].z;
            Bts[buf][(bc4[u] + 3) * LDT + bkr[u]] = (__bf16)bReg[u].w;
        }
    };

    v8f acc[2][4];
#pragma unroll
    for (int i = 0; i < 2; ++i)
#pragma unroll
        for (int j = 0; j < 4; ++j) acc[i][j] = (v8f){};

    load_regs(0);
    store_tiles(0);
    const int nK = Hv / KSTEP;
    for (int kt = 0; kt < nK; ++kt) {
        const int cur = kt & 1;
        __syncthreads();
        if (kt + 1 < nK) load_regs((kt + 1) * KSTEP);
        mma_step(As[cur], Bts[cur], wm, wn, lane, acc);
        if (kt + 1 < nK) store_tiles(1 - cur);
    }

    const float qscale = 0.03125f;  // 1/sqrt(1024)
#pragma unroll
    for (int i = 0; i < 2; ++i)
#pragma unroll
        for (int j = 0; j < 4; ++j)
#pragma unroll
            for (int r = 0; r < 8; ++r) {
                int mg = m0 + wm + i * 16 + r + ((lane & 16) ? 8 : 0);
                int ng = n0 + wn + j * 16 + (lane & 15);
                float v = acc[i][j][r];
                if (ng < Hv) v *= qscale;
                M[(size_t)mg * H3v + ng] = (__bf16)v;
            }
}

// ---------------------------------------------------------------------------
// Kernel 2: scores S[b,i,j] = Q[b,i,:].K[b,j,:]  (bf16 out).
// Lower-triangular block grid.  Q/K tiles fetched with async global->LDS
// copies, double-buffered, one barrier per K-step.
// ---------------------------------------------------------------------------
__global__ __launch_bounds__(256) void qk_scores_kernel(
    const __bf16* __restrict__ qkv, __bf16* __restrict__ P) {
    const int i0 = blockIdx.x * TILE;
    const int j0 = blockIdx.y * TILE;
    if (j0 > i0) return;                 // fully-masked causal block
    const int b  = blockIdx.z;
    __shared__ __bf16 Qs[2][TILE * LDT];
    __shared__ __bf16 Ks[2][TILE * LDT];
    const int tid  = threadIdx.x;
    const int lane = tid & 31, wave = tid >> 5;
    const int wm = (wave & 3) * 32;
    const int wn = (wave >> 2) * 64;
    const __bf16* Qbase = qkv + (size_t)b * Sv * H3v + (size_t)i0 * H3v;       // Q rows
    const __bf16* Kbase = qkv + (size_t)b * Sv * H3v + (size_t)j0 * H3v + Hv;  // K rows

    const int crow = tid >> 2;           // 0..63 (per u: +64)
    const int cc8  = (tid & 3) * 8;
    auto issue = [&](int buf, int k0) {
#pragma unroll
        for (int u = 0; u < 2; ++u) {
            int row = u * 64 + crow;
            unsigned voff = (unsigned)(row * (H3v * 2) + (k0 + cc8) * 2);
            async_copy_b128(&Qs[buf][row * LDT + cc8], Qbase, voff);
            async_copy_b128(&Ks[buf][row * LDT + cc8], Kbase, voff);
        }
    };

    v8f acc[2][4];
#pragma unroll
    for (int i = 0; i < 2; ++i)
#pragma unroll
        for (int j = 0; j < 4; ++j) acc[i][j] = (v8f){};

    issue(0, 0);
    const int nK = Hv / KSTEP;
    for (int kt = 0; kt < nK; ++kt) {
        const int cur = kt & 1;
        wait_async0();
        __syncthreads();
        if (kt + 1 < nK) issue(1 - cur, (kt + 1) * KSTEP);
        mma_step(Qs[cur], Ks[cur], wm, wn, lane, acc);
    }

    __bf16* Pb = P + (size_t)b * Sv * Sv;
#pragma unroll
    for (int i = 0; i < 2; ++i)
#pragma unroll
        for (int j = 0; j < 4; ++j)
#pragma unroll
            for (int r = 0; r < 8; ++r) {
                int ig = i0 + wm + i * 16 + r + ((lane & 16) ? 8 : 0);
                int jg = j0 + wn + j * 16 + (lane & 15);
                Pb[(size_t)ig * Sv + jg] = (__bf16)acc[i][j][r];
            }
}

// ---------------------------------------------------------------------------
// Kernel 3: causal row softmax, in place on P.  One block per (b, i) row.
// Zeros the masked tail of the diagonal 128-block for the PV GEMM.
// ---------------------------------------------------------------------------
__global__ __launch_bounds__(256) void softmax_kernel(__bf16* __restrict__ P) {
    const int i = blockIdx.x;
    const int b = blockIdx.y;
    const int tid = threadIdx.x;
    __bf16* row = P + ((size_t)b * Sv + i) * Sv;
    const int n = i + 1;
    const int kLimit = ((i >> 7) + 1) << 7;
    __shared__ float red[256];

    float m = -INFINITY;
    for (int j = tid; j < n; j += 256) m = fmaxf(m, (float)row[j]);
    red[tid] = m;
    __syncthreads();
    for (int s = 128; s > 0; s >>= 1) {
        if (tid < s) red[tid] = fmaxf(red[tid], red[tid + s]);
        __syncthreads();
    }
    m = red[0];
    __syncthreads();

    float sum = 0.0f;
    for (int j = tid; j < n; j += 256) sum += __expf((float)row[j] - m);
    red[tid] = sum;
    __syncthreads();
    for (int s = 128; s > 0; s >>= 1) {
        if (tid < s) red[tid] += red[tid + s];
        __syncthreads();
    }
    const float inv = 1.0f / red[0];

    for (int j = tid; j < kLimit; j += 256) {
        float p = (j < n) ? __expf((float)row[j] - m) * inv : 0.0f;
        row[j] = (__bf16)p;
    }
}

// ---------------------------------------------------------------------------
// Kernel 4: O[b,i,h] = sum_j P[b,i,j] * V[b,j,h]  (fp32 out).
// k-loop clipped at the diagonal block.  P tile via async global->LDS;
// V tile transposed through registers; double-buffered.
// ---------------------------------------------------------------------------
__global__ __launch_bounds__(256) void pv_gemm_kernel(
    const __bf16* __restrict__ qkv, const __bf16* __restrict__ P,
    float* __restrict__ out) {
    const int b  = blockIdx.z;
    const int i0 = blockIdx.x * TILE;     // query rows
    const int n0 = blockIdx.y * TILE;     // h columns
    __shared__ __bf16 Ps[2][TILE * LDT];  // [i][j]
    __shared__ __bf16 Vts[2][TILE * LDT]; // [h][j]
    const int tid  = threadIdx.x;
    const int lane = tid & 31, wave = tid >> 5;
    const int wm = (wave & 3) * 32;
    const int wn = (wave >> 2) * 64;
    const __bf16* Pbase = P + (size_t)b * Sv * Sv + (size_t)i0 * Sv;
    const __bf16* Vg = qkv + (size_t)b * Sv * H3v + 2 * Hv;   // V slice
    const int kTiles = (i0 >> 5) + 4;

    const int prow = tid >> 2;            // P tile coords (0..63, +64 for u=1)
    const int pc8  = (tid & 3) * 8;
    auto issueP = [&](int buf, int k0) {
#pragma unroll
        for (int u = 0; u < 2; ++u) {
            int row = u * 64 + prow;
            unsigned voff = (unsigned)(row * (Sv * 2) + (k0 + pc8) * 2);
            async_copy_b128(&Ps[buf][row * LDT + pc8], Pbase, voff);
        }
    };
    const int vkr = tid >> 4;             // V tile coords (0..15, +16 for u=1)
    const int vc8 = (tid & 15) * 8;
    auto loadV = [&](int buf, int k0) {
#pragma unroll
        for (int u = 0; u < 2; ++u) {
            int krow = u * 16 + vkr;
            U8pack v;
            v.q = *(const uint4*)(Vg + (size_t)(k0 + krow) * H3v + n0 + vc8);
#pragma unroll
            for (int j = 0; j < 8; ++j) Vts[buf][(vc8 + j) * LDT + krow] = v.h[j];
        }
    };

    v8f acc[2][4];
#pragma unroll
    for (int i = 0; i < 2; ++i)
#pragma unroll
        for (int j = 0; j < 4; ++j) acc[i][j] = (v8f){};

    issueP(0, 0);
    loadV(0, 0);
    for (int kt = 0; kt < kTiles; ++kt) {
        const int cur = kt & 1;
        wait_async0();
        __syncthreads();
        if (kt + 1 < kTiles) {
            issueP(1 - cur, (kt + 1) * KSTEP);
            loadV(1 - cur, (kt + 1) * KSTEP);
        }
        mma_step(Ps[cur], Vts[cur], wm, wn, lane, acc);
    }

#pragma unroll
    for (int i = 0; i < 2; ++i)
#pragma unroll
        for (int j = 0; j < 4; ++j)
#pragma unroll
            for (int r = 0; r < 8; ++r) {
                int ig = i0 + wm + i * 16 + r + ((lane & 16) ? 8 : 0);
                int ng = n0 + wn + j * 16 + (lane & 15);
                out[((size_t)b * Sv + ig) * Hv + ng] = acc[i][j][r];
            }
}

// ---------------------------------------------------------------------------
// Host launcher.  Workspace: QKV bf16 (48 MB) + P bf16 (33.5 MB) ~ 84 MB.
// ---------------------------------------------------------------------------
extern "C" void kernel_launch(void* const* d_in, const int* in_sizes, int n_in,
                              void* d_out, int out_size, void* d_ws, size_t ws_size,
                              hipStream_t stream) {
    (void)in_sizes; (void)n_in; (void)out_size; (void)ws_size;
    const float* x = (const float*)d_in[0];      // [B,S,H] fp32
    const float* W = (const float*)d_in[1];      // [H,3H] fp32
    float* out = (float*)d_out;                  // [B,S,H] fp32

    __bf16* qkv = (__bf16*)d_ws;
    __bf16* P   = (__bf16*)((char*)d_ws + (size_t)MTOT * H3v * sizeof(__bf16));

    dim3 blk(256);
    qkv_gemm_kernel<<<dim3(MTOT / TILE, H3v / TILE), blk, 0, stream>>>(x, W, qkv);
    qk_scores_kernel<<<dim3(Sv / TILE, Sv / TILE, Bv), blk, 0, stream>>>(qkv, P);
    softmax_kernel<<<dim3(Sv, Bv), blk, 0, stream>>>(P);
    pv_gemm_kernel<<<dim3(Sv / TILE, Hv / TILE, Bv), blk, 0, stream>>>(qkv, P, out);
}
